// HGTAttention_79499844649035
// MI455X (gfx1250) — matrix operand order repaired and checked by reading, not compiled, not run on hardware
//
#include <hip/hip_runtime.h>

// ---------------------------------------------------------------------------
// HGT hetero window attention for MI455X (gfx1250, wave32, WMMA 16x16x32 f16)
// Pipeline: prep(f16 weights) -> qkv proj -> fused attention -> out proj.
// All GEMM stages use v_wmma_f32_16x16x32_f16 with f32 accumulation.
// B operands are staged K-contiguous so every fragment is b128 LDS loads.
// f16 LDS staging (attention + out-projection) uses the CDNA5 async DMA path
// (global_load_async_to_lds_b128 / ASYNCcnt).
// Workspace requirement: ~77.4 MB (4x f16 token buffers + f16 weights).
// ---------------------------------------------------------------------------

typedef __attribute__((ext_vector_type(16))) _Float16 v16h;
typedef __attribute__((ext_vector_type(8)))  _Float16 v8h;
typedef __attribute__((ext_vector_type(8)))  float    v8f;

#define AGENTS 6
#define WIN    7
#define PTOK   49      // tokens per (agent, window)
#define HEADS  8
#define CH     32
#define DIM    256
#define NWIN   64      // X*Y
#define BATCH  2
#define NT     2

static __device__ inline v8f zero8() {
  v8f z = {0.f, 0.f, 0.f, 0.f, 0.f, 0.f, 0.f, 0.f};
  return z;
}

static __device__ inline v8f wmma_f16(v16h a, v16h b, v8f c) {
  // (neg_a, A, neg_b, B, c_mod, C, reuse_a, reuse_b)
  return __builtin_amdgcn_wmma_f32_16x16x32_f16(false, a, false, b, (short)0, c,
                                                false, false);
}

// Async global->LDS 16-byte copy (GVS mode: SGPR base + per-lane u32 offset).
// Tracked by ASYNCcnt; fence with async_wait0() before the LDS is consumed.
static __device__ inline void async_cp16(unsigned lds_off, unsigned gofs,
                                         unsigned long long gbase) {
  asm volatile("global_load_async_to_lds_b128 %0, %1, %2 offset:0"
               :: "v"(lds_off), "v"(gofs), "s"(gbase) : "memory");
}
static __device__ inline void async_wait0() {
  asm volatile("s_wait_asynccnt 0" ::: "memory");
}

// A fragment (16x32 f16): lane lo holds row (row0+lo); elements are
// K=16*hi+e (contiguous 16 halves, matching frag_bc).
static __device__ inline v16h frag_a(const _Float16* base, int row0, int stride,
                                     int k0) {
  const int lane = threadIdx.x & 31;
  const int lo = lane & 15, hi = lane >> 4;
  const _Float16* r = base + (size_t)(row0 + lo) * stride + k0 + 16 * hi;
  return *(const v16h*)r;
}

// B fragment (32x16 f16): lane lo holds one column; elements are K=16*hi+e.
// `col` must point at this lane's column stored K-contiguous (32 halves).
static __device__ inline v16h frag_bc(const _Float16* col) {
  const int hi = (threadIdx.x >> 4) & 1;
  return *(const v16h*)(col + 16 * hi);
}

// ---------------------------------------------------------------------------
// Kernel 0: convert rel_att / rel_msg / Wo to f16 once.
// ---------------------------------------------------------------------------
__global__ __launch_bounds__(256) void prep_kernel(
    const float* __restrict__ ra, const float* __restrict__ rm,
    const float* __restrict__ Wo, _Float16* __restrict__ wa_h,
    _Float16* __restrict__ wm_h, _Float16* __restrict__ wo_h) {
  const int idx = blockIdx.x * 256 + threadIdx.x;
  if (idx < NT * NT * HEADS * CH * CH) {   // 32768
    wa_h[idx] = (_Float16)ra[idx];
    wm_h[idx] = (_Float16)rm[idx];
  }
  if (idx < DIM * DIM) wo_h[idx] = (_Float16)Wo[idx];
}

// ---------------------------------------------------------------------------
// Kernel 1: QKV projection.  Per (b,l): [3136,256] x [256,256] GEMM for q,k,v
// sharing the x-block in LDS.  Output f16, layout [b][n][m][l][p][c], q scaled.
// grid = (49 Mtiles, 4 Ntiles, 12 (b,l)); block = 256 (8 waves x 2 tiles).
// ---------------------------------------------------------------------------
__global__ __launch_bounds__(256) void proj_kernel(
    const float* __restrict__ x, const float* __restrict__ Wq,
    const float* __restrict__ bq, const float* __restrict__ Wk,
    const float* __restrict__ bk, const float* __restrict__ Wv,
    const float* __restrict__ bv, const int* __restrict__ agent_type,
    _Float16* __restrict__ qbuf, _Float16* __restrict__ kbuf,
    _Float16* __restrict__ vbuf) {
  extern __shared__ char smem[];
  _Float16* sA = (_Float16*)smem;    // [64][256] x block (f16)
  _Float16* sB = sA + 64 * DIM;      // [64 col][32 k]  W sub-block, transposed

  const int mt_blk = blockIdx.x;     // 0..48
  const int nt_blk = blockIdx.y;     // 0..3
  const int bz = blockIdx.z;         // 0..11
  const int b = bz / AGENTS, l = bz % AGENTS;
  const int tid = threadIdx.x, wave = tid >> 5, lane = tid & 31;
  const int lo = lane & 15, hi = lane >> 4;
  const int t_l = (l == 0) ? 0 : *agent_type;
  const float scale = 0.17677669529663687f;  // 32^-0.5

  // Load A block (64 tokens x 256), f32 -> f16.
  const float* xsrc =
      x + ((size_t)bz * (NWIN * PTOK) + (size_t)mt_blk * 64) * DIM;
  for (int idx = tid; idx < 64 * DIM; idx += 256)
    sA[idx] = (_Float16)xsrc[idx];
  __syncthreads();

  const float* Wsel[3] = {Wq + (size_t)t_l * DIM * DIM,
                          Wk + (size_t)t_l * DIM * DIM,
                          Wv + (size_t)t_l * DIM * DIM};
  const float* bsel[3] = {bq + t_l * DIM, bk + t_l * DIM, bv + t_l * DIM};
  _Float16* osel[3] = {qbuf, kbuf, vbuf};

  for (int o = 0; o < 3; ++o) {
    const float* W = Wsel[o];
    v8f acc[2];
    acc[0] = zero8();
    acc[1] = zero8();
    for (int kb = 0; kb < 8; ++kb) {
      __syncthreads();
      // prefetch next K block of W into L2
      if (kb + 1 < 8 && tid < 32)
        __builtin_prefetch(W + (size_t)((kb + 1) * 32 + tid) * DIM +
                               nt_blk * 64, 0, 0);
      // stage W sub-block transposed: sB[col][k]  (coalesced global reads)
      for (int idx = tid; idx < 32 * 64; idx += 256) {
        const int kk = idx >> 6, ee = idx & 63;
        sB[ee * 32 + kk] =
            (_Float16)W[(size_t)(kb * 32 + kk) * DIM + nt_blk * 64 + ee];
      }
      __syncthreads();
#pragma unroll
      for (int t = 0; t < 2; ++t) {
        const int id = wave * 2 + t, mt = id >> 2, nn = id & 3;
        v16h a = frag_a(sA, mt * 16, DIM, kb * 32);
        v16h bf = frag_bc(sB + (nn * 16 + lo) * 32);
        acc[t] = wmma_f16(a, bf, acc[t]);
      }
    }
#pragma unroll
    for (int t = 0; t < 2; ++t) {
      const int id = wave * 2 + t, mt = id >> 2, nn = id & 3;
      const int e = nt_blk * 64 + nn * 16 + lo;
      const int m = e >> 5, c = e & 31;
      const float bval = bsel[o][e];
#pragma unroll
      for (int r = 0; r < 8; ++r) {
        const int trow = mt_blk * 64 + mt * 16 + r + 8 * hi;
        const int n = trow / PTOK, p = trow % PTOK;
        float v = acc[t][r] + bval;
        if (o == 0) v *= scale;
        osel[o][(((((size_t)b * NWIN + n) * HEADS + m) * AGENTS + l) * PTOK +
                 p) * CH + c] = (_Float16)v;
      }
    }
  }
}

// ---------------------------------------------------------------------------
// Kernel 2: fused hetero attention.  One WG per (b, window, head, agent i).
// grid = (64, 8, 12); block = 256 (8 waves).  131072 B dynamic LDS.
// f16 staging via global_load_async_to_lds_b128 (ASYNCcnt fenced).
// ---------------------------------------------------------------------------
__global__ __launch_bounds__(256) void attn_kernel(
    const _Float16* __restrict__ qbuf, const _Float16* __restrict__ kbuf,
    const _Float16* __restrict__ vbuf, const _Float16* __restrict__ wa_h,
    const _Float16* __restrict__ wm_h, const float* __restrict__ bias_table,
    const int* __restrict__ agent_type, _Float16* __restrict__ abuf) {
  extern __shared__ char smem[];
  _Float16* sQ = (_Float16*)smem;        // [64][32]
  _Float16* sK = sQ + 64 * 32;           // [6][64][32]  (later reused as sVTt)
  _Float16* sV = sK + 6 * 64 * 32;       // [6][64][32]
  _Float16* sWaT = sV + 6 * 64 * 32;     // [6][32(f)][32(c)]  w_att transposed
  _Float16* sWm = sWaT + 6 * 32 * 32;    // [6][32(c)][32(f)]
  _Float16* sQW = sWm + 6 * 32 * 32;     // [64][32]
  _Float16* sSim = sQW + 64 * 32;        // [64][384]  logits -> attn probs

  const int n = blockIdx.x, m = blockIdx.y;
  const int b = blockIdx.z / AGENTS, i = blockIdx.z % AGENTS;
  const int tid = threadIdx.x, wave = tid >> 5, lane = tid & 31;
  const int lo = lane & 15, hi = lane >> 4;
  const int atype = *agent_type;
  const int ti = (i == 0) ? 0 : atype;

  const size_t head_base = (((size_t)b * NWIN + n) * HEADS + m) * AGENTS;
  const uint4 z4 = make_uint4(0u, 0u, 0u, 0u);

  const unsigned ldsQ = (unsigned)(unsigned long long)(void*)sQ;
  const unsigned ldsK = (unsigned)(unsigned long long)(void*)sK;
  const unsigned ldsV = (unsigned)(unsigned long long)(void*)sV;
  const unsigned ldsWm = (unsigned)(unsigned long long)(void*)sWm;
  const unsigned long long gQ = (unsigned long long)(const void*)qbuf;
  const unsigned long long gK = (unsigned long long)(const void*)kbuf;
  const unsigned long long gV = (unsigned long long)(const void*)vbuf;
  const unsigned long long gWm = (unsigned long long)(const void*)wm_h;

  // ---- stage LDS: async DMA for valid rows, ds_store zero for pad rows.
  for (int idx = tid; idx < 64 * 4; idx += 256) {
    const int row = idx >> 2, ck = idx & 3;
    if (row < PTOK)
      async_cp16(ldsQ + (unsigned)idx * 16u,
                 (unsigned)((((head_base + i) * PTOK + row) * CH + ck * 8) * 2),
                 gQ);
    else
      *(uint4*)(sQ + row * 32 + ck * 8) = z4;
  }
  for (int idx = tid; idx < 6 * 64 * 4; idx += 256) {
    const int rr = idx >> 2, ck = idx & 3;
    const int j = rr >> 6, qq = rr & 63;
    if (qq < PTOK) {
      const unsigned gofs =
          (unsigned)((((head_base + j) * PTOK + qq) * CH + ck * 8) * 2);
      async_cp16(ldsK + (unsigned)idx * 16u, gofs, gK);
      async_cp16(ldsV + (unsigned)idx * 16u, gofs, gV);
    } else {
      *(uint4*)(sK + rr * 32 + ck * 8) = z4;
      *(uint4*)(sV + rr * 32 + ck * 8) = z4;
    }
  }
  // w_msg as-is (async b128); w_att transposed through VGPRs (reshape needed).
  for (int idx = tid; idx < 6 * 128; idx += 256) {
    const int j = idx >> 7, rem = idx & 127;
    const int tj = (j == 0) ? 0 : atype;
    const size_t wbase = (((size_t)(ti * NT + tj) * HEADS + m) << 10);
    async_cp16(ldsWm + (unsigned)idx * 16u,
               (unsigned)((wbase + rem * 8) * 2), gWm);
  }
  for (int idx = tid; idx < 6 * 32 * 32; idx += 256) {
    const int j = idx >> 10, rem = idx & 1023;
    const int tj = (j == 0) ? 0 : atype;
    const int f = rem >> 5, c = rem & 31;
    const size_t wbase = (((size_t)(ti * NT + tj) * HEADS + m) << 10);
    sWaT[idx] = wa_h[wbase + (c << 5) + f];   // [j][f][c]
  }
  async_wait0();
  __syncthreads();

  // ---- sim = (q_i @ w_att[i][j]) @ k_j^T  + rel-pos bias, per agent j
  for (int j = 0; j < 6; ++j) {
    {  // qw block: 8 tiles (4M x 2N), one per wave, K = 32 (c dim)
      const int mt = wave >> 1, nn = wave & 1;
      v16h a = frag_a(sQ, mt * 16, 32, 0);
      v16h bf = frag_bc(sWaT + (j << 10) + ((nn * 16 + lo) << 5));
      v8f acc = zero8();
      acc = wmma_f16(a, bf, acc);
#pragma unroll
      for (int r = 0; r < 8; ++r)
        sQW[((mt * 16 + r + 8 * hi) << 5) + nn * 16 + lo] = (_Float16)acc[r];
    }
    __syncthreads();
    // sim block: 16 tiles (4M x 4N), 2 per wave, K = 32 (f dim)
    for (int t = 0; t < 2; ++t) {
      const int id = wave * 2 + t, mt = id >> 2, nn = id & 3;
      v16h a = frag_a(sQW, mt * 16, 32, 0);
      const int qc = nn * 16 + lo;
      v16h bf = frag_bc(sK + (size_t)(j << 6) * 32 + (qc << 5));
      v8f acc = zero8();
      acc = wmma_f16(a, bf, acc);
#pragma unroll
      for (int r = 0; r < 8; ++r) {
        const int p = mt * 16 + r + 8 * hi;
        float v = -65504.f;  // mask padded rows/cols
        if (p < PTOK && qc < PTOK) {
          const int relidx = ((i - j + AGENTS - 1) * 13 +
                              (p / WIN - qc / WIN + WIN - 1)) * 13 +
                             (p % WIN - qc % WIN + WIN - 1);
          v = acc[r] + bias_table[relidx * HEADS + m];
        }
        sSim[p * 384 + (j << 6) + qc] = (_Float16)v;
      }
    }
    __syncthreads();
  }

  // ---- softmax over 384 cols, 4 lanes per row, wave32 shfl_xor reductions
  {
    const int row = tid >> 2, quad = tid & 3;
    _Float16* rp = sSim + row * 384;
    const int c0 = quad * 96, c1 = c0 + 96;
    float mx = -3.0e38f;
    for (int c = c0; c < c1; ++c) mx = fmaxf(mx, (float)rp[c]);
    mx = fmaxf(mx, __shfl_xor(mx, 1, 32));
    mx = fmaxf(mx, __shfl_xor(mx, 2, 32));
    float sum = 0.f;
    for (int c = c0; c < c1; ++c) {
      const float e = __expf((float)rp[c] - mx);
      rp[c] = (_Float16)e;
      sum += e;
    }
    sum += __shfl_xor(sum, 1, 32);
    sum += __shfl_xor(sum, 2, 32);
    const float inv = 1.f / sum;
    for (int c = c0; c < c1; ++c) rp[c] = (_Float16)((float)rp[c] * inv);
  }

  // ---- vt[j] = v_j @ w_msg[i][j]^T, stored TRANSPOSED over sK: sVTt[32][384]
  _Float16* sVTt = sK;  // [c=32][qg=384]
  {
    const int mt = wave >> 1, nn = wave & 1;
    for (int j = 0; j < 6; ++j) {
      v16h a = frag_a(sV + (size_t)(j << 6) * 32, mt * 16, 32, 0);
      const int c = nn * 16 + lo;
      v16h bf = frag_bc(sWm + (j << 10) + (c << 5));
      v8f acc = zero8();
      acc = wmma_f16(a, bf, acc);
#pragma unroll
      for (int r = 0; r < 8; ++r) {
        const int qg = (j << 6) + mt * 16 + r + 8 * hi;
        sVTt[c * 384 + qg] = (_Float16)acc[r];
      }
    }
  }
  __syncthreads();

  // ---- out = attn[64,384] @ vt[384,32]; 8 tiles, one per wave, 12 K-steps
  {
    const int mt = wave >> 1, nn = wave & 1;
    const int c = nn * 16 + lo;
    v8f acc = zero8();
    for (int kb = 0; kb < 12; ++kb) {
      v16h a = frag_a(sSim, mt * 16, 384, kb * 32);
      v16h bf = frag_bc(sVTt + c * 384 + kb * 32);
      acc = wmma_f16(a, bf, acc);
    }
#pragma unroll
    for (int r = 0; r < 8; ++r) {
      const int p = mt * 16 + r + 8 * hi;
      if (p < PTOK)
        abuf[((((size_t)b * AGENTS + i) * NWIN + n) * PTOK + p) * DIM +
             m * CH + c] = (_Float16)acc[r];
    }
  }
}

// ---------------------------------------------------------------------------
// Kernel 3: output projection [37632,256] @ Wo^T -> f32 out.
// grid = (588 Mtiles, 4 Ntiles); block = 256.  B column dd == Wo row dd.
// LDS staging uses gfx1250 async copies (ASYNCcnt) instead of VGPR round-trip.
// ---------------------------------------------------------------------------
__global__ __launch_bounds__(256) void outproj_kernel(
    const _Float16* __restrict__ abuf, const _Float16* __restrict__ wo_h,
    float* __restrict__ out) {
  extern __shared__ char smem[];
  _Float16* sA = (_Float16*)smem;  // [64][256]
  _Float16* sB = sA + 64 * DIM;    // [64(dd)][256(e)]  (K-contiguous columns)

  const int mt_blk = blockIdx.x, nt_blk = blockIdx.y;
  const int tid = threadIdx.x, wave = tid >> 5, lane = tid & 31;
  const int lo = lane & 15, hi = lane >> 4;

  // Async global -> LDS staging of both 32 KB tiles (b128 per lane).
  {
    const unsigned long long gA =
        (unsigned long long)(const void*)(abuf + (size_t)mt_blk * 64 * DIM);
    const unsigned long long gB =
        (unsigned long long)(const void*)(wo_h + (size_t)nt_blk * 64 * DIM);
    const unsigned ldsA = (unsigned)(unsigned long long)(void*)sA;
    const unsigned ldsB = (unsigned)(unsigned long long)(void*)sB;
    for (int idx = tid; idx < 2048; idx += 256) {
      const unsigned off = (unsigned)idx * 16u;
      async_cp16(ldsA + off, off, gA);
      async_cp16(ldsB + off, off, gB);
    }
    async_wait0();
  }
  __syncthreads();

  v8f acc[2];
  acc[0] = zero8();
  acc[1] = zero8();
  for (int kb = 0; kb < 8; ++kb) {
#pragma unroll
    for (int t = 0; t < 2; ++t) {
      const int id = wave * 2 + t, mt = id >> 2, nn = id & 3;
      v16h a = frag_a(sA, mt * 16, DIM, kb * 32);
      v16h bf = frag_bc(sB + (nn * 16 + lo) * DIM + kb * 32);
      acc[t] = wmma_f16(a, bf, acc[t]);
    }
  }
#pragma unroll
  for (int t = 0; t < 2; ++t) {
    const int id = wave * 2 + t, mt = id >> 2, nn = id & 3;
    const int col = nt_blk * 64 + nn * 16 + lo;
#pragma unroll
    for (int r = 0; r < 8; ++r) {
      const int trow = mt_blk * 64 + mt * 16 + r + 8 * hi;
      out[(size_t)trow * DIM + col] = acc[t][r];
    }
  }
}

// ---------------------------------------------------------------------------
extern "C" void kernel_launch(void* const* d_in, const int* in_sizes, int n_in,
                              void* d_out, int out_size, void* d_ws,
                              size_t ws_size, hipStream_t stream) {
  (void)in_sizes; (void)n_in; (void)out_size; (void)ws_size;

  const float* x  = (const float*)d_in[0];
  const float* Wq = (const float*)d_in[1];
  const float* bq = (const float*)d_in[2];
  const float* Wk = (const float*)d_in[3];
  const float* bk = (const float*)d_in[4];
  const float* Wv = (const float*)d_in[5];
  const float* bv = (const float*)d_in[6];
  const float* ra = (const float*)d_in[7];
  const float* rm = (const float*)d_in[8];
  const float* bt = (const float*)d_in[9];
  const float* Wo = (const float*)d_in[10];
  const int* atype = (const int*)d_in[11];

  const size_t QKV = (size_t)BATCH * NWIN * HEADS * AGENTS * PTOK * CH; // 9.63M
  _Float16* qbuf = (_Float16*)d_ws;
  _Float16* kbuf = qbuf + QKV;
  _Float16* vbuf = kbuf + QKV;
  _Float16* abuf = vbuf + QKV;           // attention output, pre-Wo (f16)
  _Float16* wa_h = abuf + QKV;           // 32768
  _Float16* wm_h = wa_h + 32768;         // 32768
  _Float16* wo_h = wm_h + 32768;         // 65536

  prep_kernel<<<256, 256, 0, stream>>>(ra, rm, Wo, wa_h, wm_h, wo_h);

  dim3 g1(49, 4, BATCH * AGENTS);
  proj_kernel<<<g1, 256, (64 * DIM + 64 * 32) * sizeof(_Float16), stream>>>(
      x, Wq, bq, Wk, bk, Wv, bv, atype, qbuf, kbuf, vbuf);

  dim3 g2(NWIN, HEADS, BATCH * AGENTS);
  attn_kernel<<<g2, 256, 131072, stream>>>(qbuf, kbuf, vbuf, wa_h, wm_h, bt,
                                           atype, abuf);

  dim3 g3(588, 4, 1);
  outproj_kernel<<<g3, 256, 2 * 64 * DIM * sizeof(_Float16), stream>>>(
      abuf, wo_h, (float*)d_out);
}
